// PointTransformer2D_3D_PETR_49435073577587
// MI455X (gfx1250) — matrix operand order repaired and checked by reading, not compiled, not run on hardware
//
#include <hip/hip_runtime.h>
#include <math.h>

// ---------------------------------------------------------------------------
// Model constants
// ---------------------------------------------------------------------------
#define C_DIM   256
#define P_DIM   200
#define S_DIM   32400
#define FFN_DIM 1024
#define BN_SCALE 0.9999950000374997f    // 1/sqrt(1 + 1e-5)
#define LOG_EPS32 (-15.942385f)         // logf(FLT_EPSILON)
#define INV_SQRT_D 0.17677669529663687f // 1/sqrt(32)

#if __has_builtin(__builtin_amdgcn_global_load_async_to_lds_b128)
#define HAVE_ASYNC_LDS 1
#else
#define HAVE_ASYNC_LDS 0
#endif

typedef __attribute__((ext_vector_type(4))) int i32x4;

__device__ __forceinline__ void wait_asynccnt0() {
#if __has_builtin(__builtin_amdgcn_s_wait_asynccnt)
  __builtin_amdgcn_s_wait_asynccnt(0);
#else
  asm volatile("s_wait_asynccnt 0x0" ::: "memory");
#endif
}

// ---------------------------------------------------------------------------
// WMMA types (CDNA5 / gfx1250, wave32)
// ---------------------------------------------------------------------------
typedef __attribute__((ext_vector_type(16))) __bf16 bf16x16;
typedef __attribute__((ext_vector_type(8)))  float  f32x8;

union BF16x16 { unsigned short u[16]; bf16x16 v; };

__device__ __forceinline__ unsigned short f2bf(float x) {
  union { float f; unsigned int u; } cv; cv.f = x;
  unsigned int b = cv.u;
  b += 0x7fffu + ((b >> 16) & 1u);   // round-to-nearest-even
  return (unsigned short)(b >> 16);
}

// ---------------------------------------------------------------------------
// Generic WMMA GEMM:  C[m,n] = f( alpha * sum_k A(m,k)*B(k,n) + biases )
//   a_kmajor=1 : A at A[m*lda+k]      a_kmajor=0 : A at A[k*lda+m]
//   b_nk=1     : B at B[n*ldb+k]      b_nk=0     : B at B[k*ldb+n]
// Block = 256 threads = 8 waves; block tile 64(M) x 128(N); K-step 32.
// Each wave owns a 32x32 macro-tile -> 4 v_wmma_f32_16x16x32_bf16 per K-step.
// fp32 -> bf16 conversion happens during LDS staging (one cvt per element).
// ---------------------------------------------------------------------------
#define BM 64
#define BN 128
#define LDSA_STRIDE 40
#define LDSB_STRIDE 40

__device__ __forceinline__ BF16x16 load_frag_a(const unsigned short* tile, int lane) {
  // 16-bit A 16x32: lanes 0-15 K{0..7,16..23}, lanes 16-31 K{8..15,24..31}
  int row = lane & 15;
  int kh  = (lane >> 4) << 3;
  const unsigned short* p = tile + row * LDSA_STRIDE + kh;
  BF16x16 f;
#pragma unroll
  for (int i = 0; i < 8; ++i) f.u[i]     = p[i];
#pragma unroll
  for (int i = 0; i < 8; ++i) f.u[8 + i] = p[16 + i];
  return f;
}

__device__ __forceinline__ BF16x16 load_frag_b(const unsigned short* tile, int lane) {
  // B 32x16: lane holds column n=lane&15; lanes 0-15 K{0..7,16..23}, lanes 16-31 K{8..15,24..31}
  int col = lane & 15;
  int kh  = (lane >> 4) << 3;
  const unsigned short* p = tile + col * LDSB_STRIDE + kh;
  BF16x16 f;
#pragma unroll
  for (int i = 0; i < 8; ++i) f.u[i]     = p[i];
#pragma unroll
  for (int i = 0; i < 8; ++i) f.u[8 + i] = p[16 + i];
  return f;
}

__device__ __forceinline__ void store_tile(float* __restrict__ Cp, const f32x8& acc,
                                           int m_base, int n_base, int M, int N, int ldc,
                                           float alpha,
                                           const float* __restrict__ bias_m,
                                           const float* __restrict__ bias_n,
                                           const float* __restrict__ bn_g,
                                           const float* __restrict__ bn_be,
                                           int relu, int lane)
{
  int n  = n_base + (lane & 15);
  int mb = m_base + ((lane >> 4) * 8);
  if (n >= N) return;
  float bnv = bias_n ? bias_n[n] : 0.0f;
#pragma unroll
  for (int r = 0; r < 8; ++r) {
    int m = mb + r;
    if (m < M) {
      float v = acc[r] * alpha;
      if (bias_m) v += bias_m[m];
      if (bn_g)   v  = v * (bn_g[m] * BN_SCALE) + bn_be[m];
      v += bnv;
      if (relu)   v  = fmaxf(v, 0.0f);
      Cp[(size_t)m * ldc + n] = v;
    }
  }
}

__global__ __launch_bounds__(256)
void wmma_gemm_kernel(const float* __restrict__ A, const float* __restrict__ B,
                      float* __restrict__ Cp,
                      int M, int N, int K, int lda, int ldb, int ldc,
                      int a_kmajor, int b_nk, float alpha,
                      const float* __restrict__ bias_m,
                      const float* __restrict__ bias_n,
                      const float* __restrict__ bn_g,
                      const float* __restrict__ bn_be,
                      int relu)
{
  __shared__ unsigned short As[BM * LDSA_STRIDE];
  __shared__ unsigned short Bs[BN * LDSB_STRIDE];

  const int tid  = threadIdx.x;
  const int lane = tid & 31;
  const int w    = tid >> 5;
  const int wm   = w >> 2;          // 0..1
  const int wn   = w & 3;           // 0..3
  const int m0   = blockIdx.y * BM;
  const int n0   = blockIdx.x * BN;

  f32x8 acc00 = {}, acc01 = {}, acc10 = {}, acc11 = {};

  const bool interior = (m0 + BM <= M) && (n0 + BN <= N);

  for (int k0 = 0; k0 < K; k0 += 32) {
    __syncthreads();
    if (interior && (k0 + 32 <= K)) {
      // ---------------- fast path: unguarded, vectorized ----------------
      {
        // A tile 64x32: 8 consecutive floats per thread
        int r = tid >> 2;
        int c = (tid & 3) * 8;
        unsigned short* dst = &As[r * LDSA_STRIDE + c];
        if (a_kmajor) {
          const float4* ga = (const float4*)&A[(size_t)(m0 + r) * lda + (k0 + c)];
          float4 v0 = ga[0], v1 = ga[1];
          dst[0] = f2bf(v0.x); dst[1] = f2bf(v0.y); dst[2] = f2bf(v0.z); dst[3] = f2bf(v0.w);
          dst[4] = f2bf(v1.x); dst[5] = f2bf(v1.y); dst[6] = f2bf(v1.z); dst[7] = f2bf(v1.w);
        } else {
#pragma unroll
          for (int j = 0; j < 8; ++j)
            dst[j] = f2bf(A[(size_t)(k0 + c + j) * lda + (m0 + r)]);
        }
      }
      {
        // B tile 128x32: 16 consecutive floats per thread
        int n = tid >> 1;
        int c = (tid & 1) * 16;
        unsigned short* dst = &Bs[n * LDSB_STRIDE + c];
        if (b_nk) {
          const float4* gb = (const float4*)&B[(size_t)(n0 + n) * ldb + (k0 + c)];
          float4 v0 = gb[0], v1 = gb[1], v2 = gb[2], v3 = gb[3];
          dst[0]  = f2bf(v0.x); dst[1]  = f2bf(v0.y); dst[2]  = f2bf(v0.z); dst[3]  = f2bf(v0.w);
          dst[4]  = f2bf(v1.x); dst[5]  = f2bf(v1.y); dst[6]  = f2bf(v1.z); dst[7]  = f2bf(v1.w);
          dst[8]  = f2bf(v2.x); dst[9]  = f2bf(v2.y); dst[10] = f2bf(v2.z); dst[11] = f2bf(v2.w);
          dst[12] = f2bf(v3.x); dst[13] = f2bf(v3.y); dst[14] = f2bf(v3.z); dst[15] = f2bf(v3.w);
        } else {
#pragma unroll
          for (int j = 0; j < 16; ++j)
            dst[j] = f2bf(B[(size_t)(k0 + c + j) * ldb + (n0 + n)]);
        }
      }
    } else {
      // ---------------- edge path: per-element guarded ----------------
#pragma unroll
      for (int t = 0; t < 8; ++t) {
        int idx = tid + t * 256;
        int r = idx >> 5, c = idx & 31;
        int gm = m0 + r, gk = k0 + c;
        float v = 0.0f;
        if (gm < M && gk < K)
          v = a_kmajor ? A[(size_t)gm * lda + gk] : A[(size_t)gk * lda + gm];
        As[r * LDSA_STRIDE + c] = f2bf(v);
      }
#pragma unroll
      for (int t = 0; t < 16; ++t) {
        int idx = tid + t * 256;
        int n = idx >> 5, c = idx & 31;
        int gn = n0 + n, gk = k0 + c;
        float v = 0.0f;
        if (gn < N && gk < K)
          v = b_nk ? B[(size_t)gn * ldb + gk] : B[(size_t)gk * ldb + gn];
        Bs[n * LDSB_STRIDE + c] = f2bf(v);
      }
    }
    // speculative prefetch of the next A K-tile (global_prefetch_b8)
    if (k0 + 32 < K) {
      int r = tid >> 5, c = tid & 31;
      int gm = m0 + r;      if (gm >= M) gm = M - 1;
      int gk = k0 + 32 + c; if (gk >= K) gk = K - 1;
      const float* pa = a_kmajor ? &A[(size_t)gm * lda + gk]
                                 : &A[(size_t)gk * lda + gm];
      __builtin_prefetch(pa, 0, 1);
    }
    __syncthreads();

    BF16x16 fa0 = load_frag_a(&As[(wm * 32 +  0) * LDSA_STRIDE], lane);
    BF16x16 fa1 = load_frag_a(&As[(wm * 32 + 16) * LDSA_STRIDE], lane);
    BF16x16 fb0 = load_frag_b(&Bs[(wn * 32 +  0) * LDSB_STRIDE], lane);
    BF16x16 fb1 = load_frag_b(&Bs[(wn * 32 + 16) * LDSB_STRIDE], lane);

    acc00 = __builtin_amdgcn_wmma_f32_16x16x32_bf16(false, fa0.v, false, fb0.v, (short)0, acc00, false, false);
    acc01 = __builtin_amdgcn_wmma_f32_16x16x32_bf16(false, fa0.v, false, fb1.v, (short)0, acc01, false, false);
    acc10 = __builtin_amdgcn_wmma_f32_16x16x32_bf16(false, fa1.v, false, fb0.v, (short)0, acc10, false, false);
    acc11 = __builtin_amdgcn_wmma_f32_16x16x32_bf16(false, fa1.v, false, fb1.v, (short)0, acc11, false, false);
  }

  const int mb0 = m0 + wm * 32;
  const int nb0 = n0 + wn * 32;
  store_tile(Cp, acc00, mb0,      nb0,      M, N, ldc, alpha, bias_m, bias_n, bn_g, bn_be, relu, lane);
  store_tile(Cp, acc01, mb0,      nb0 + 16, M, N, ldc, alpha, bias_m, bias_n, bn_g, bn_be, relu, lane);
  store_tile(Cp, acc10, mb0 + 16, nb0,      M, N, ldc, alpha, bias_m, bias_n, bn_g, bn_be, relu, lane);
  store_tile(Cp, acc11, mb0 + 16, nb0 + 16, M, N, ldc, alpha, bias_m, bias_n, bn_g, bn_be, relu, lane);
}

// ---------------------------------------------------------------------------
// Position-embed first conv + BN + ReLU:  out[c,n] = relu(bn(W1[c,:D]·pos[n,:D]+b1))
// ---------------------------------------------------------------------------
__global__ void pe_first_kernel(const float* __restrict__ pos, int N, int D,
                                const float* __restrict__ w1,
                                const float* __restrict__ b1,
                                const float* __restrict__ g,
                                const float* __restrict__ be,
                                float* __restrict__ out)
{
  int i = blockIdx.x * blockDim.x + threadIdx.x;
  if (i >= C_DIM * N) return;
  int n = i % N, c = i / N;
  float s = b1[c];
  for (int d = 0; d < D; ++d) s += w1[c * D + d] * pos[n * D + d];
  s = s * (g[c] * BN_SCALE) + be[c];
  out[(size_t)c * N + n] = fmaxf(s, 0.0f);
}

// kv[s,c] += lidar[c,s]   (kv is [S,C] row-major, lidar is [C,S])
__global__ void kv_add_kernel(float* __restrict__ kv, const float* __restrict__ lidar)
{
  int i = blockIdx.x * blockDim.x + threadIdx.x;
  if (i >= S_DIM * C_DIM) return;
  int s = i / C_DIM, c = i % C_DIM;
  kv[i] += lidar[(size_t)c * S_DIM + s];
}

// out[p,c] = a[c,p] + b[c,p]
__global__ void addT_kernel(float* __restrict__ out, const float* __restrict__ a,
                            const float* __restrict__ b)
{
  int i = blockIdx.x * blockDim.x + threadIdx.x;
  if (i >= P_DIM * C_DIM) return;
  int p = i / C_DIM, c = i % C_DIM;
  out[i] = a[(size_t)c * P_DIM + p] + b[(size_t)c * P_DIM + p];
}

// out[p,c] = a[p,c] + b[c,p]
__global__ void add_pT_kernel(float* __restrict__ out, const float* __restrict__ a,
                              const float* __restrict__ b)
{
  int i = blockIdx.x * blockDim.x + threadIdx.x;
  if (i >= P_DIM * C_DIM) return;
  int p = i / C_DIM, c = i % C_DIM;
  out[i] = a[i] + b[(size_t)c * P_DIM + p];
}

// ---------------------------------------------------------------------------
// Row softmax with optional fused SMCA gaussian log-bias.
// One block (256 threads) per row p. The row is cached in (dynamic) LDS via
// CDNA5 async global->LDS copies (ASYNCcnt), then max/exp/scale run from LDS.
// dynamic smem = (Lk + 256) floats.  Lk must be a multiple of 4 (200 / 32400).
// ---------------------------------------------------------------------------
__global__ void softmax_kernel(float* __restrict__ scr, int Lk, int lds,
                               const float* __restrict__ centerb,   // [2,P] or null
                               const float* __restrict__ dimb,      // [3,P]
                               const float* __restrict__ bev)       // [S,2]
{
  extern __shared__ float smrow[];             // Lk floats row cache
  float* red = smrow + Lk;                     // 256 floats reduction scratch

  const int p = blockIdx.x;
  const int tid = threadIdx.x;
  float* row = scr + (size_t)p * lds;

  // ---- stage row into LDS ----
#if HAVE_ASYNC_LDS
  for (int q = tid; q < (Lk >> 2); q += 256) {
    __builtin_amdgcn_global_load_async_to_lds_b128(
        (__attribute__((address_space(1))) i32x4*)(row + q * 4),
        (__attribute__((address_space(3))) i32x4*)(smrow + q * 4), 0, 0);
  }
  wait_asynccnt0();
  __syncthreads();
#else
  for (int q = tid; q < Lk; q += 256) smrow[q] = row[q];
  __syncthreads();
#endif

  float cx = 0.f, cy = 0.f, inv2s2 = 0.f;
  const bool hasb = (centerb != nullptr);
  if (hasb) {
    cx = centerb[p]; cy = centerb[P_DIM + p];
    float ex = expf(dimb[p])         * (1.0f / 0.6f);   // /(8*0.075)
    float ey = expf(dimb[P_DIM + p]) * (1.0f / 0.6f);
    float radius = ceilf(sqrtf(ex * ex + ey * ey) * 0.5f);
    float sigma  = (radius * 2.0f + 1.0f) * (1.0f / 6.0f);
    inv2s2 = 1.0f / (2.0f * sigma * sigma);
  }

  float mx = -INFINITY;
  for (int s = tid; s < Lk; s += 256) {
    float v = smrow[s];
    if (hasb) {
      float dx = cx - (bev[2 * s]     - 0.5f);
      float dy = cy - (bev[2 * s + 1] - 0.5f);
      float t = -(dx * dx + dy * dy) * inv2s2;
      v += (t >= LOG_EPS32) ? t : -INFINITY;
    }
    mx = fmaxf(mx, v);
  }
  red[tid] = mx; __syncthreads();
  for (int st = 128; st > 0; st >>= 1) {
    if (tid < st) red[tid] = fmaxf(red[tid], red[tid + st]);
    __syncthreads();
  }
  mx = red[0]; __syncthreads();
  if (mx == -INFINITY) {
    for (int s = tid; s < Lk; s += 256) row[s] = 0.0f;
    return;
  }
  float sum = 0.0f;
  for (int s = tid; s < Lk; s += 256) {
    float v = smrow[s];
    if (hasb) {
      float dx = cx - (bev[2 * s]     - 0.5f);
      float dy = cy - (bev[2 * s + 1] - 0.5f);
      float t = -(dx * dx + dy * dy) * inv2s2;
      v += (t >= LOG_EPS32) ? t : -INFINITY;
    }
    float e = expf(v - mx);
    smrow[s] = e;
    sum += e;
  }
  red[tid] = sum; __syncthreads();
  for (int st = 128; st > 0; st >>= 1) {
    if (tid < st) red[tid] += red[tid + st];
    __syncthreads();
  }
  float inv = 1.0f / red[0];
  for (int s = tid; s < Lk; s += 256) row[s] = smrow[s] * inv;
}

// ---------------------------------------------------------------------------
// Residual + LayerNorm over C=256.  x may be [C,P] (x_t=1) or [P,C]; r is [P,C].
// ---------------------------------------------------------------------------
__global__ void ln_kernel(const float* __restrict__ x, int x_t, int ldx,
                          const float* __restrict__ r,
                          const float* __restrict__ g, const float* __restrict__ b,
                          float* __restrict__ y, float* __restrict__ yT)
{
  const int p = blockIdx.x, c = threadIdx.x;
  __shared__ float red[256];
  float xv = x_t ? x[(size_t)c * ldx + p] : x[(size_t)p * ldx + c];
  float s = xv + r[p * C_DIM + c];
  red[c] = s; __syncthreads();
  for (int st = 128; st > 0; st >>= 1) { if (c < st) red[c] += red[c + st]; __syncthreads(); }
  float mean = red[0] * (1.0f / C_DIM); __syncthreads();
  float d = s - mean;
  red[c] = d * d; __syncthreads();
  for (int st = 128; st > 0; st >>= 1) { if (c < st) red[c] += red[c + st]; __syncthreads(); }
  float var = red[0] * (1.0f / C_DIM);
  float yv = d * rsqrtf(var + 1e-5f) * g[c] + b[c];
  y[p * C_DIM + c] = yv;
  if (yT) yT[(size_t)c * P_DIM + p] = yv;
}

// ---------------------------------------------------------------------------
// Small utility kernels
// ---------------------------------------------------------------------------
__global__ void copy_kernel(float* __restrict__ dst, const float* __restrict__ src, int n)
{
  int i = blockIdx.x * blockDim.x + threadIdx.x;
  if (i < n) dst[i] = src[i];
}

__global__ void qpos_init_kernel(float* __restrict__ qpos, const float* __restrict__ pos2d,
                                 const float* __restrict__ initg)
{
  int p = blockIdx.x * blockDim.x + threadIdx.x;
  if (p >= P_DIM) return;
  qpos[p * 3 + 0] = pos2d[p * 2 + 0];
  qpos[p * 3 + 1] = pos2d[p * 2 + 1];
  qpos[p * 3 + 2] = initg[p];
}

__global__ void center_init_kernel(float* __restrict__ centerb, const float* __restrict__ pos2d)
{
  int p = blockIdx.x * blockDim.x + threadIdx.x;
  if (p >= P_DIM) return;
  centerb[p]         = pos2d[p * 2 + 0];
  centerb[P_DIM + p] = pos2d[p * 2 + 1];
}

__global__ void resid_update_kernel(const float* __restrict__ ch, const float* __restrict__ hh,
                                    float* __restrict__ qpos,
                                    float* __restrict__ centerb, float* __restrict__ heightb,
                                    float* __restrict__ out_center, float* __restrict__ out_height)
{
  int p = blockIdx.x * blockDim.x + threadIdx.x;
  if (p >= P_DIM) return;
  float cx = ch[p]         + qpos[p * 3 + 0];
  float cy = ch[P_DIM + p] + qpos[p * 3 + 1];
  float hz = hh[p]         + qpos[p * 3 + 2];
  centerb[p] = cx; centerb[P_DIM + p] = cy; heightb[p] = hz;
  qpos[p * 3 + 0] = cx; qpos[p * 3 + 1] = cy; qpos[p * 3 + 2] = hz;
  out_center[p] = cx; out_center[P_DIM + p] = cy;
  out_height[p] = hz;
}

// ---------------------------------------------------------------------------
// Host-side orchestration
// ---------------------------------------------------------------------------
struct HeadP  { const float *b1, *b2, *be, *g, *w1, *w2; };
struct MhaP   { const float *in_b, *in_w, *out_b, *out_w; };
struct PeP    { const float *b1, *b2, *be, *g, *w1, *w2; };
struct LayerP {
  MhaP ca; PeP cpe;
  const float *l1b, *l1w, *l2b, *l2w;
  const float *n1b, *n1g, *n2b, *n2g, *n3b, *n3g;
  MhaP sa; PeP spe;
};

static inline void gemm(hipStream_t st, const float* A, const float* B, float* Cp,
                        int M, int N, int K, int lda, int ldb, int ldc,
                        int a_kmajor, int b_nk, float alpha,
                        const float* bm, const float* bn_, const float* g,
                        const float* be, int relu)
{
  dim3 grid((N + BN - 1) / BN, (M + BM - 1) / BM);
  hipLaunchKernelGGL(wmma_gemm_kernel, grid, dim3(256), 0, st,
                     A, B, Cp, M, N, K, lda, ldb, ldc,
                     a_kmajor, b_nk, alpha, bm, bn_, g, be, relu);
}

static inline void run_head(hipStream_t st, const float* qf, const HeadP& H,
                            int cout, float* out, float* h1ws)
{
  gemm(st, H.w1, qf, h1ws, 64, P_DIM, C_DIM, C_DIM, P_DIM, P_DIM,
       1, 0, 1.0f, H.b1, nullptr, H.g, H.be, 1);
  gemm(st, H.w2, h1ws, out, cout, P_DIM, 64, 64, P_DIM, P_DIM,
       1, 0, 1.0f, H.b2, nullptr, nullptr, nullptr, 0);
}

extern "C" void kernel_launch(void* const* d_in, const int* in_sizes, int n_in,
                              void* d_out, int out_size, void* d_ws, size_t ws_size,
                              hipStream_t stream)
{
  if (n_in < 98) return;  // expects jax-tree flattened param order

  const float* qfeat = (const float*)d_in[0];   // [C,P]
  const float* pos2d = (const float*)d_in[1];   // [P,2]
  const float* lidar = (const float*)d_in[2];   // [C,S]
  const float* bev   = (const float*)d_in[3];   // [S,2]

  auto F = [&](int i) { return (const float*)d_in[i]; };
  int ix = 4;
  HeadP hd[5];  // alphabetical: center, dim, heatmap, height, rot
  for (int h = 0; h < 5; ++h) {
    hd[h].b1 = F(ix++); hd[h].b2 = F(ix++); hd[h].be = F(ix++);
    hd[h].g  = F(ix++); hd[h].w1 = F(ix++); hd[h].w2 = F(ix++);
  }
  const HeadP& hd_center = hd[0];
  const HeadP& hd_dim    = hd[1];
  const HeadP& hd_heat   = hd[2];
  const HeadP& hd_height = hd[3];
  const HeadP& hd_rot    = hd[4];
  const float* ih_b1 = F(ix++); const float* ih_b2 = F(ix++);
  const float* ih_w1 = F(ix++); const float* ih_w2 = F(ix++);
  LayerP L[2];
  for (int l = 0; l < 2; ++l) {
    LayerP& LY = L[l];
    LY.ca.in_b = F(ix++); LY.ca.in_w = F(ix++); LY.ca.out_b = F(ix++); LY.ca.out_w = F(ix++);
    LY.cpe.b1 = F(ix++); LY.cpe.b2 = F(ix++); LY.cpe.be = F(ix++);
    LY.cpe.g  = F(ix++); LY.cpe.w1 = F(ix++); LY.cpe.w2 = F(ix++);
    LY.l1b = F(ix++); LY.l1w = F(ix++); LY.l2b = F(ix++); LY.l2w = F(ix++);
    LY.n1b = F(ix++); LY.n1g = F(ix++); LY.n2b = F(ix++); LY.n2g = F(ix++);
    LY.n3b = F(ix++); LY.n3g = F(ix++);
    LY.sa.in_b = F(ix++); LY.sa.in_w = F(ix++); LY.sa.out_b = F(ix++); LY.sa.out_w = F(ix++);
    LY.spe.b1 = F(ix++); LY.spe.b2 = F(ix++); LY.spe.be = F(ix++);
    LY.spe.g  = F(ix++); LY.spe.w1 = F(ix++); LY.spe.w2 = F(ix++);
  }

  // ---------- workspace carving (fp32) ----------
  float* ws = (float*)d_ws;
  size_t off = 0;
  auto walloc = [&](size_t n) { float* p = ws + off; off += (n + 63) & ~(size_t)63; return p; };
  float* big0 = walloc((size_t)C_DIM * S_DIM);  // cross_pe mid / kproj
  float* big1 = walloc((size_t)S_DIM * C_DIM);  // kpe -> kv (in place)
  float* big2 = walloc((size_t)S_DIM * C_DIM);  // vproj
  float* big3 = walloc((size_t)P_DIM * S_DIM);  // cross scores / attn (in place)
  float* qpe  = walloc((size_t)C_DIM * P_DIM);
  float* t1   = walloc((size_t)C_DIM * P_DIM);
  float* xq   = walloc((size_t)P_DIM * C_DIM);
  float* sq   = walloc((size_t)P_DIM * C_DIM);
  float* sk_  = walloc((size_t)P_DIM * C_DIM);
  float* sv_  = walloc((size_t)P_DIM * C_DIM);
  float* ssc  = walloc((size_t)P_DIM * P_DIM);
  float* sout = walloc((size_t)P_DIM * C_DIM);
  float* mhab = walloc((size_t)P_DIM * C_DIM);
  float* q1   = walloc((size_t)P_DIM * C_DIM);
  float* q2   = walloc((size_t)P_DIM * C_DIM);
  float* q3   = walloc((size_t)P_DIM * C_DIM);
  float* f1   = walloc((size_t)P_DIM * FFN_DIM);
  float* qf_cur = walloc((size_t)C_DIM * P_DIM);
  float* h1   = walloc((size_t)64 * P_DIM);
  float* tmp256 = walloc((size_t)C_DIM * P_DIM);
  float* centerb = walloc(2 * P_DIM);
  float* dimb    = walloc(3 * P_DIM);
  float* heightb = walloc(P_DIM);
  float* ch   = walloc(2 * P_DIM);
  float* hh   = walloc(P_DIM);
  float* qposb = walloc(3 * P_DIM);
  float* initg = walloc(P_DIM);
  float* cq   = walloc((size_t)P_DIM * C_DIM);
  float* cout_ = walloc((size_t)P_DIM * C_DIM);
  float* outp = (float*)d_out;

  const size_t smem_self  = (size_t)(P_DIM + 256) * sizeof(float);
  const size_t smem_cross = (size_t)(S_DIM + 256) * sizeof(float);

  // ---------- init_height + qpos + initial center/dim (layer-0 bias) ----------
  gemm(stream, ih_w1, qfeat, tmp256, C_DIM, P_DIM, C_DIM, C_DIM, P_DIM, P_DIM,
       1, 0, 1.0f, ih_b1, nullptr, nullptr, nullptr, 1);
  gemm(stream, ih_w2, tmp256, initg, 1, P_DIM, C_DIM, C_DIM, P_DIM, P_DIM,
       1, 0, 1.0f, ih_b2, nullptr, nullptr, nullptr, 0);
  hipLaunchKernelGGL(qpos_init_kernel, dim3(1), dim3(256), 0, stream, qposb, pos2d, initg);
  hipLaunchKernelGGL(center_init_kernel, dim3(1), dim3(256), 0, stream, centerb, pos2d);
  run_head(stream, qfeat, hd_dim, 3, dimb, h1);

  const float* qf_in = qfeat;
  for (int i = 0; i < 2; ++i) {
    const LayerP& LY = L[i];
    float* outbase = outp + 51800 + i * 3600;   // [center|dim|heatmap|height|rot]

    // ---- self / cross position embeds ----
    hipLaunchKernelGGL(pe_first_kernel, dim3((C_DIM * P_DIM + 255) / 256), dim3(256), 0, stream,
                       qposb, P_DIM, 3, LY.spe.w1, LY.spe.b1, LY.spe.g, LY.spe.be, t1);
    gemm(stream, LY.spe.w2, t1, qpe, C_DIM, P_DIM, C_DIM, C_DIM, P_DIM, P_DIM,
         1, 0, 1.0f, LY.spe.b2, nullptr, nullptr, nullptr, 0);

    hipLaunchKernelGGL(pe_first_kernel, dim3((C_DIM * S_DIM + 255) / 256), dim3(256), 0, stream,
                       bev, S_DIM, 2, LY.cpe.w1, LY.cpe.b1, LY.cpe.g, LY.cpe.be, big0);
    gemm(stream, big0, LY.cpe.w2, big1, S_DIM, C_DIM, C_DIM, S_DIM, C_DIM, C_DIM,
         0, 1, 1.0f, nullptr, LY.cpe.b2, nullptr, nullptr, 0);
    hipLaunchKernelGGL(kv_add_kernel, dim3((S_DIM * C_DIM + 255) / 256), dim3(256), 0, stream,
                       big1, lidar);
    gemm(stream, big1, LY.ca.in_w + 256 * 256, big0, S_DIM, C_DIM, C_DIM,
         C_DIM, C_DIM, C_DIM, 1, 1, 1.0f, nullptr, LY.ca.in_b + 256, nullptr, nullptr, 0);
    gemm(stream, big1, LY.ca.in_w + 512 * 256, big2, S_DIM, C_DIM, C_DIM,
         C_DIM, C_DIM, C_DIM, 1, 1, 1.0f, nullptr, LY.ca.in_b + 512, nullptr, nullptr, 0);

    // ---- self attention ----
    hipLaunchKernelGGL(addT_kernel, dim3((P_DIM * C_DIM + 255) / 256), dim3(256), 0, stream,
                       xq, qf_in, qpe);
    gemm(stream, xq, LY.sa.in_w,             sq,  P_DIM, C_DIM, C_DIM, C_DIM, C_DIM, C_DIM,
         1, 1, 1.0f, nullptr, LY.sa.in_b,       nullptr, nullptr, 0);
    gemm(stream, xq, LY.sa.in_w + 256 * 256, sk_, P_DIM, C_DIM, C_DIM, C_DIM, C_DIM, C_DIM,
         1, 1, 1.0f, nullptr, LY.sa.in_b + 256, nullptr, nullptr, 0);
    gemm(stream, xq, LY.sa.in_w + 512 * 256, sv_, P_DIM, C_DIM, C_DIM, C_DIM, C_DIM, C_DIM,
         1, 1, 1.0f, nullptr, LY.sa.in_b + 512, nullptr, nullptr, 0);
    for (int h = 0; h < 8; ++h) {
      gemm(stream, sq + h * 32, sk_ + h * 32, ssc, P_DIM, P_DIM, 32,
           C_DIM, C_DIM, P_DIM, 1, 1, INV_SQRT_D, nullptr, nullptr, nullptr, nullptr, 0);
      hipLaunchKernelGGL(softmax_kernel, dim3(P_DIM), dim3(256), smem_self, stream,
                         ssc, P_DIM, P_DIM, (const float*)nullptr, (const float*)nullptr,
                         (const float*)nullptr);
      gemm(stream, ssc, sv_ + h * 32, sout + h * 32, P_DIM, 32, P_DIM,
           P_DIM, C_DIM, C_DIM, 1, 0, 1.0f, nullptr, nullptr, nullptr, nullptr, 0);
    }
    gemm(stream, sout, LY.sa.out_w, mhab, P_DIM, C_DIM, C_DIM, C_DIM, C_DIM, C_DIM,
         1, 1, 1.0f, nullptr, LY.sa.out_b, nullptr, nullptr, 0);
    hipLaunchKernelGGL(ln_kernel, dim3(P_DIM), dim3(256), 0, stream,
                       qf_in, 1, P_DIM, mhab, LY.n1g, LY.n1b, q1, (float*)nullptr);

    // ---- cross attention (SMCA bias fused in softmax) ----
    hipLaunchKernelGGL(add_pT_kernel, dim3((P_DIM * C_DIM + 255) / 256), dim3(256), 0, stream,
                       xq, q1, qpe);
    gemm(stream, xq, LY.ca.in_w, cq, P_DIM, C_DIM, C_DIM, C_DIM, C_DIM, C_DIM,
         1, 1, 1.0f, nullptr, LY.ca.in_b, nullptr, nullptr, 0);
    for (int h = 0; h < 8; ++h) {
      gemm(stream, cq + h * 32, big0 + h * 32, big3, P_DIM, S_DIM, 32,
           C_DIM, C_DIM, S_DIM, 1, 1, INV_SQRT_D, nullptr, nullptr, nullptr, nullptr, 0);
      hipLaunchKernelGGL(softmax_kernel, dim3(P_DIM), dim3(256), smem_cross, stream,
                         big3, S_DIM, S_DIM, centerb, dimb, bev);
      gemm(stream, big3, big2 + h * 32, cout_ + h * 32, P_DIM, 32, S_DIM,
           S_DIM, C_DIM, C_DIM, 1, 0, 1.0f, nullptr, nullptr, nullptr, nullptr, 0);
    }
    gemm(stream, cout_, LY.ca.out_w, mhab, P_DIM, C_DIM, C_DIM, C_DIM, C_DIM, C_DIM,
         1, 1, 1.0f, nullptr, LY.ca.out_b, nullptr, nullptr, 0);
    hipLaunchKernelGGL(ln_kernel, dim3(P_DIM), dim3(256), 0, stream,
                       q1, 0, C_DIM, mhab, LY.n2g, LY.n2b, q2, (float*)nullptr);

    // ---- FFN ----
    gemm(stream, q2, LY.l1w, f1, P_DIM, FFN_DIM, C_DIM, C_DIM, C_DIM, FFN_DIM,
         1, 1, 1.0f, nullptr, LY.l1b, nullptr, nullptr, 1);
    gemm(stream, f1, LY.l2w, mhab, P_DIM, C_DIM, FFN_DIM, FFN_DIM, FFN_DIM, C_DIM,
         1, 1, 1.0f, nullptr, LY.l2b, nullptr, nullptr, 0);
    hipLaunchKernelGGL(ln_kernel, dim3(P_DIM), dim3(256), 0, stream,
                       q2, 0, C_DIM, mhab, LY.n3g, LY.n3b, q3, qf_cur);

    // ---- prediction heads ----
    run_head(stream, qf_cur, hd_center, 2, ch, h1);
    run_head(stream, qf_cur, hd_height, 1, hh, h1);
    run_head(stream, qf_cur, hd_dim,    3, dimb, h1);
    run_head(stream, qf_cur, hd_rot,    2, outbase + 3200, h1);
    run_head(stream, qf_cur, hd_heat,  10, outbase + 1000, h1);
    hipLaunchKernelGGL(copy_kernel, dim3(3), dim3(256), 0, stream, outbase + 400, dimb, 600);
    hipLaunchKernelGGL(resid_update_kernel, dim3(1), dim3(256), 0, stream,
                       ch, hh, qposb, centerb, heightb, outbase, outbase + 3000);

    qf_in = qf_cur;
  }

  // ---- final outputs: qf [C,P], qpos [P,3] ----
  hipLaunchKernelGGL(copy_kernel, dim3((51200 + 255) / 256), dim3(256), 0, stream,
                     outp, qf_cur, 51200);
  hipLaunchKernelGGL(copy_kernel, dim3(3), dim3(256), 0, stream,
                     outp + 51200, qposb, 600);
  (void)in_sizes; (void)out_size; (void)ws_size;
}
